// PointTransformerBlock_34488587387649
// MI455X (gfx1250) — compile-verified, matrix operand order checked by reading, not run on hardware
//
#include <hip/hip_runtime.h>
#include <hip/hip_bf16.h>
#include <math.h>

#define BATCH 2
#define NPTS  4096
#define DIM   256
#define KNN   16
#define MTOT  (BATCH * NPTS)   // 8192 points
#define PTS_PER_WAVE 4

typedef __bf16 bf16;
typedef __attribute__((ext_vector_type(16))) __bf16        v16bf;
typedef __attribute__((ext_vector_type(8)))  float         v8f;
typedef __attribute__((ext_vector_type(8)))  unsigned int  v8u;
typedef __attribute__((ext_vector_type(4)))  unsigned int  u32x4;
typedef __attribute__((ext_vector_type(8)))  int           i32x8;
typedef __attribute__((ext_vector_type(4)))  int           i32x4;

union Frag { v8u u; v16bf bf; };

// K offset within a 16-bit A/B fragment, per ISA 16-bit 16x32 layout:
// VGPR v in [0,8): lanes 0-15 hold K = base..base+7, lanes 16-31 hold +8.
__device__ __forceinline__ int koff(int v, int half) {
  return ((v >> 2) << 4) + half * 8 + ((v & 3) << 1);
}

__device__ __forceinline__ v8f wmma_bf16(const Frag& a, const Frag& b, v8f c) {
  return __builtin_amdgcn_wmma_f32_16x16x32_bf16(
      /*neg_a=*/false, a.bf, /*neg_b=*/false, b.bf,
      /*c_mod=*/(short)0, c, /*reuse_a=*/false, /*reuse_b=*/false);
}

// ---------------------------------------------------------------------------
// Prep: f32 -> bf16 conversion and weight transpose (store Wt[n*256+k])
// ---------------------------------------------------------------------------
__global__ void cvt_bf16_k(const float* __restrict__ in, bf16* __restrict__ out, int n) {
  int i = blockIdx.x * blockDim.x + threadIdx.x;
  if (i < n) out[i] = (bf16)in[i];
}

__global__ void transpose_bf16_k(const float* __restrict__ w, bf16* __restrict__ wt) {
  int k = blockIdx.x, n = threadIdx.x;                 // 256 x 256
  wt[n * DIM + k] = (bf16)w[k * DIM + n];
}

// ---------------------------------------------------------------------------
// KNN: one thread per query point, pos tiles staged in LDS, register top-16
// ---------------------------------------------------------------------------
__global__ __launch_bounds__(256) void knn_k(const float* __restrict__ pos,
                                             int* __restrict__ knn_out) {
  __shared__ float sp[256][3];
  const int tid = threadIdx.x;
  const int p = blockIdx.x * 256 + tid;        // global point id
  const int b = blockIdx.x >> 4;               // 16 blocks per batch (4096/256)

  const float qx = pos[(size_t)p * 3 + 0];
  const float qy = pos[(size_t)p * 3 + 1];
  const float qz = pos[(size_t)p * 3 + 2];

  float bd[KNN];
  int   bi[KNN];
#pragma unroll
  for (int t = 0; t < KNN; ++t) { bd[t] = 3.0e38f; bi[t] = 0; }

  for (int tile = 0; tile < NPTS / 256; ++tile) {
    int j0 = b * NPTS + tile * 256;
    sp[tid][0] = pos[(size_t)(j0 + tid) * 3 + 0];
    sp[tid][1] = pos[(size_t)(j0 + tid) * 3 + 1];
    sp[tid][2] = pos[(size_t)(j0 + tid) * 3 + 2];
    __syncthreads();
    for (int jj = 0; jj < 256; ++jj) {
      float dx = qx - sp[jj][0], dy = qy - sp[jj][1], dz = qz - sp[jj][2];
      float d = dx * dx + dy * dy + dz * dz;
      if (d < bd[KNN - 1]) {
        bd[KNN - 1] = d; bi[KNN - 1] = tile * 256 + jj;
#pragma unroll
        for (int t = KNN - 1; t > 0; --t) {
          if (bd[t] < bd[t - 1]) {
            float td = bd[t]; bd[t] = bd[t - 1]; bd[t - 1] = td;
            int   ti = bi[t]; bi[t] = bi[t - 1]; bi[t - 1] = ti;
          }
        }
      }
    }
    __syncthreads();
  }
#pragma unroll
  for (int t = 0; t < KNN; ++t) knn_out[(size_t)p * KNN + t] = bi[t];
}

// ---------------------------------------------------------------------------
// Tiled WMMA GEMM: C[M x 256] = A[M x 256] @ W + bias (+resid). A,Wt bf16,
// Wt column-major (Wt[n*256+k]). One 16x16 tile per wave, 4 waves/block.
// ---------------------------------------------------------------------------
__global__ __launch_bounds__(128) void gemm_wmma_bf16_k(
    const bf16* __restrict__ A, const bf16* __restrict__ Wt,
    const float* __restrict__ bias, const float* __restrict__ resid,
    bf16* __restrict__ outb, float* __restrict__ outf) {
  const int lane = threadIdx.x & 31;
  const int wave = threadIdx.x >> 5;
  const int tile = blockIdx.x * 4 + wave;
  const int m0 = (tile >> 4) * 16;
  const int n0 = (tile & 15) * 16;
  const int nl = lane & 15, half = lane >> 4;

  float binit = bias ? bias[n0 + nl] : 0.0f;
  v8f acc;
#pragma unroll
  for (int r = 0; r < 8; ++r) acc[r] = binit;

  const bf16* arow = A + (size_t)(m0 + nl) * DIM;
  const bf16* brow = Wt + (size_t)(n0 + nl) * DIM;
#pragma unroll
  for (int s = 0; s < 8; ++s) {
    Frag af, bfg;
#pragma unroll
    for (int v = 0; v < 8; ++v) {
      af.u[v]  = *(const unsigned int*)(arow + s * 32 + koff(v, half));
      bfg.u[v] = *(const unsigned int*)(brow + s * 32 + koff(v, half));
    }
    acc = wmma_bf16(af, bfg, acc);
  }
#pragma unroll
  for (int r = 0; r < 8; ++r) {
    int row = m0 + r + half * 8;
    int col = n0 + nl;
    float v = acc[r];
    if (resid) v += resid[(size_t)row * DIM + col];
    if (outf) outf[(size_t)row * DIM + col] = v;
    if (outb) outb[(size_t)row * DIM + col] = (bf16)v;
  }
}

// ---------------------------------------------------------------------------
// Stage a 256x256 bf16 weight matrix (128 KB) into LDS via the Tensor Data
// Mover; falls back to a cooperative vector copy if the builtin is absent.
// ---------------------------------------------------------------------------
__device__ __forceinline__ void stage_weight(bf16* sW, const bf16* __restrict__ Wt,
                                             int lane) {
#if defined(__gfx1250__) && __has_builtin(__builtin_amdgcn_tensor_load_to_lds)
  unsigned lds_addr = (unsigned)(size_t)(void*)sW;   // low 32 bits = LDS offset
  unsigned long long ga = (unsigned long long)(size_t)(const void*)Wt;
  // D# group0: count=1, lds_addr, global_addr[56:0], type=2 ("image")
  u32x4 g0 = { 1u, lds_addr,
               (unsigned)(ga & 0xFFFFFFFFu),
               (unsigned)((ga >> 32) & 0x01FFFFFFu) | (2u << 30) };
  // D# group1: data_size=8B; tensor_dim0=tile_dim0=16384; dim1=1; stride0=16384
  i32x8 g1 = { 0x00030000,   // mask=0, data_size=3 (8B)
               0x40000000,   // tensor_dim0 lo16 (0x4000) << 16
               0x00010000,   // tensor_dim0 hi16=0 | tensor_dim1 lo16 = 1
               0x40000000,   // tensor_dim1 hi16=0 | tile_dim0 = 0x4000
               0x00000001,   // tile_dim1 = 1, tile_dim2 = 0
               0x00004000,   // tensor_dim0_stride lo32 = 16384
               0x40000000,   // stride0 hi16=0 | tensor_dim1_stride lo16 = 0x4000
               0x00000000 }; // tensor_dim1_stride hi32 = 0
  i32x4 gz = { 0, 0, 0, 0 };
#if __clang_major__ >= 23
  i32x8 gz8 = { 0, 0, 0, 0, 0, 0, 0, 0 };
  __builtin_amdgcn_tensor_load_to_lds(g0, g1, gz, gz, gz8, 0);
#else
  __builtin_amdgcn_tensor_load_to_lds(g0, g1, gz, gz, 0);
#endif
#if __has_builtin(__builtin_amdgcn_s_wait_tensorcnt)
  __builtin_amdgcn_s_wait_tensorcnt(0);
#endif
#else
  const uint4* src = (const uint4*)Wt;
  uint4* dst = (uint4*)sW;
  for (int i = lane; i < (DIM * DIM * 2) / 16; i += 32) dst[i] = src[i];
#endif
  __syncthreads();
}

// ---------------------------------------------------------------------------
// Per-wave MLP GEMM, 4 points stacked: O[4][16 x 256] = act(A @ W + bias).
// A in LDS (bf16 row-major per point), W in LDS (bf16 column-major).
// Each B fragment feeds 4 WMMAs -> 4x weight-read amortization.
// ---------------------------------------------------------------------------
__device__ __forceinline__ void wave_mlp_gemm4(const bf16* sA, const bf16* sW,
                                               const float* __restrict__ bias,
                                               bool relu, bf16* sOut, int lane) {
  const int nl = lane & 15, half = lane >> 4;
  for (int nt = 0; nt < 16; ++nt) {
    float binit = bias[nt * 16 + nl];
    v8f acc[PTS_PER_WAVE];
#pragma unroll
    for (int pt = 0; pt < PTS_PER_WAVE; ++pt)
#pragma unroll
      for (int r = 0; r < 8; ++r) acc[pt][r] = binit;

    const bf16* brow = sW + (nt * 16 + nl) * DIM;
#pragma unroll
    for (int s = 0; s < 8; ++s) {
      Frag bfg;
#pragma unroll
      for (int v = 0; v < 8; ++v)
        bfg.u[v] = *(const unsigned int*)(brow + s * 32 + koff(v, half));
#pragma unroll
      for (int pt = 0; pt < PTS_PER_WAVE; ++pt) {
        Frag af;
#pragma unroll
        for (int v = 0; v < 8; ++v)
          af.u[v] = *(const unsigned int*)(sA + pt * (KNN * DIM) + nl * DIM +
                                           s * 32 + koff(v, half));
        acc[pt] = wmma_bf16(af, bfg, acc[pt]);
      }
    }
#pragma unroll
    for (int pt = 0; pt < PTS_PER_WAVE; ++pt)
#pragma unroll
      for (int r = 0; r < 8; ++r) {
        float v = acc[pt][r];
        if (relu) v = fmaxf(v, 0.0f);
        sOut[pt * (KNN * DIM) + (r + half * 8) * DIM + nt * 16 + nl] = (bf16)v;
      }
  }
}

// ---------------------------------------------------------------------------
// Fused attention: one wave per 4 points. Weights TDM-staged in LDS, three
// WMMA MLP phases, softmax over K neighbors, aggregation -> aggb (bf16).
// ---------------------------------------------------------------------------
__global__ __launch_bounds__(32) void attn_k(
    const float* __restrict__ pos, const bf16* __restrict__ qb,
    const bf16* __restrict__ kb, const bf16* __restrict__ vb,
    const int* __restrict__ knn_in,
    const float* __restrict__ pw1, const float* __restrict__ pb1,
    const bf16* __restrict__ pw2T, const float* __restrict__ pb2,
    const bf16* __restrict__ aw1T, const float* __restrict__ ab1,
    const bf16* __restrict__ aw2T, const float* __restrict__ ab2,
    bf16* __restrict__ aggb) {
  extern __shared__ __align__(16) char smem_raw[];
  bf16* sain = (bf16*)smem_raw;                     // 4 x 16 x 256  (32 KB)
  bf16* spe  = sain + PTS_PER_WAVE * KNN * DIM;     // 32 KB (pos_enc)
  bf16* sact = spe  + PTS_PER_WAVE * KNN * DIM;     // 32 KB
  bf16* sW   = sact + PTS_PER_WAVE * KNN * DIM;     // 128 KB staged weight
  float* srel = (float*)(sW + DIM * DIM);           // [4*16][4]
  int*  sidx  = (int*)(srel + PTS_PER_WAVE * KNN * 4);
  bf16* saout = sain;                               // reuse after GEMM2

  const int lane = threadIdx.x;
  const int p0 = blockIdx.x * PTS_PER_WAVE;
  const int b = p0 >> 12;  // all 4 points in same batch (4096 % 4 == 0)

  // Gather neighbor indices + relative positions for 4 points (64 entries)
  for (int e = lane; e < PTS_PER_WAVE * KNN; e += 32) {
    int pt = e >> 4;
    int p = p0 + pt;
    int j = knn_in[(size_t)p * KNN + (e & 15)];
    sidx[e] = j;
    size_t cb = ((size_t)b * NPTS + j) * 3;
    size_t pb = (size_t)p * 3;
    srel[e * 4 + 0] = pos[pb + 0] - pos[cb + 0];
    srel[e * 4 + 1] = pos[pb + 1] - pos[cb + 1];
    srel[e * 4 + 2] = pos[pb + 2] - pos[cb + 2];
  }
  __syncthreads();

  // h1 = relu(rel @ pw1 + pb1) -> sain (per-lane pw1 columns preloaded)
  float pw1r[8][3], pb1r[8];
#pragma unroll
  for (int cc = 0; cc < 8; ++cc) {
    int c = lane + 32 * cc;
    pb1r[cc] = pb1[c];
#pragma unroll
    for (int d = 0; d < 3; ++d) pw1r[cc][d] = pw1[d * DIM + c];
  }
  for (int e = 0; e < PTS_PER_WAVE * KNN; ++e) {
    float rx = srel[e * 4 + 0], ry = srel[e * 4 + 1], rz = srel[e * 4 + 2];
#pragma unroll
    for (int cc = 0; cc < 8; ++cc) {
      float h = pb1r[cc] + rx * pw1r[cc][0] + ry * pw1r[cc][1] + rz * pw1r[cc][2];
      sain[e * DIM + lane + 32 * cc] = (bf16)fmaxf(h, 0.0f);
    }
  }
  __syncthreads();

  stage_weight(sW, pw2T, lane);
  wave_mlp_gemm4(sain, sW, pb2, false, spe, lane);   // pos_enc
  __syncthreads();

  // a_in = q - k_gathered + pos_enc -> sain
  for (int pt = 0; pt < PTS_PER_WAVE; ++pt) {
    int p = p0 + pt;
    float qreg[8];
#pragma unroll
    for (int cc = 0; cc < 8; ++cc)
      qreg[cc] = (float)qb[(size_t)p * DIM + lane + 32 * cc];
    for (int m = 0; m < KNN; ++m) {
      int e = pt * KNN + m;
      size_t kbase = ((size_t)b * NPTS + sidx[e]) * DIM;
#pragma unroll
      for (int cc = 0; cc < 8; ++cc) {
        int c = lane + 32 * cc;
        float v = qreg[cc] - (float)kb[kbase + c] + (float)spe[e * DIM + c];
        sain[e * DIM + c] = (bf16)v;
      }
    }
  }
  __syncthreads();

  stage_weight(sW, aw1T, lane);
  wave_mlp_gemm4(sain, sW, ab1, true, sact, lane);
  __syncthreads();
  stage_weight(sW, aw2T, lane);
  wave_mlp_gemm4(sact, sW, ab2, false, saout, lane);
  __syncthreads();

  // softmax over K per channel, then aggregate attn * (v + pos_enc)
  for (int pt = 0; pt < PTS_PER_WAVE; ++pt) {
    int p = p0 + pt;
    for (int cc = 0; cc < 8; ++cc) {
      int c = lane + 32 * cc;
      float av[KNN], mx = -3.0e38f;
#pragma unroll
      for (int m = 0; m < KNN; ++m) {
        av[m] = (float)saout[(pt * KNN + m) * DIM + c] * 0.0625f;  // / sqrt(256)
        mx = fmaxf(mx, av[m]);
      }
      float e[KNN], s = 0.0f;
#pragma unroll
      for (int m = 0; m < KNN; ++m) { e[m] = __expf(av[m] - mx); s += e[m]; }
      float inv = 1.0f / s, agg = 0.0f;
#pragma unroll
      for (int m = 0; m < KNN; ++m) {
        size_t vbase = ((size_t)b * NPTS + sidx[pt * KNN + m]) * DIM;
        float vpe = (float)vb[vbase + c] + (float)spe[(pt * KNN + m) * DIM + c];
        agg += e[m] * inv * vpe;
      }
      aggb[(size_t)p * DIM + c] = (bf16)agg;
    }
  }
}

// ---------------------------------------------------------------------------
extern "C" void kernel_launch(void* const* d_in, const int* in_sizes, int n_in,
                              void* d_out, int out_size, void* d_ws, size_t ws_size,
                              hipStream_t stream) {
  (void)in_sizes; (void)n_in; (void)out_size; (void)ws_size;
  const float* x   = (const float*)d_in[0];
  const float* pos = (const float*)d_in[1];
  const float* wq  = (const float*)d_in[2];
  const float* wk  = (const float*)d_in[3];
  const float* wv  = (const float*)d_in[4];
  const float* pw1 = (const float*)d_in[5];
  const float* pb1 = (const float*)d_in[6];
  const float* pw2 = (const float*)d_in[7];
  const float* pb2 = (const float*)d_in[8];
  const float* aw1 = (const float*)d_in[9];
  const float* ab1 = (const float*)d_in[10];
  const float* aw2 = (const float*)d_in[11];
  const float* ab2 = (const float*)d_in[12];
  const float* fw  = (const float*)d_in[13];
  const float* fb  = (const float*)d_in[14];

  char* ws = (char*)d_ws;
  size_t off = 0;
  auto alloc = [&](size_t bytes) -> void* {
    void* pp = ws + off;
    off = (off + bytes + 255) & ~(size_t)255;
    return pp;
  };

  bf16* xb   = (bf16*)alloc((size_t)MTOT * DIM * 2);
  bf16* qb   = (bf16*)alloc((size_t)MTOT * DIM * 2);
  bf16* kb   = (bf16*)alloc((size_t)MTOT * DIM * 2);
  bf16* vb   = (bf16*)alloc((size_t)MTOT * DIM * 2);
  bf16* aggb = (bf16*)alloc((size_t)MTOT * DIM * 2);
  bf16* wqT  = (bf16*)alloc((size_t)DIM * DIM * 2);
  bf16* wkT  = (bf16*)alloc((size_t)DIM * DIM * 2);
  bf16* wvT  = (bf16*)alloc((size_t)DIM * DIM * 2);
  bf16* pw2T = (bf16*)alloc((size_t)DIM * DIM * 2);
  bf16* aw1T = (bf16*)alloc((size_t)DIM * DIM * 2);
  bf16* aw2T = (bf16*)alloc((size_t)DIM * DIM * 2);
  bf16* fwT  = (bf16*)alloc((size_t)DIM * DIM * 2);
  int*  knnw = (int*)alloc((size_t)MTOT * KNN * 4);

  // Prep: bf16 conversions + weight transposes
  {
    int n = MTOT * DIM;
    cvt_bf16_k<<<(n + 255) / 256, 256, 0, stream>>>(x, xb, n);
  }
  transpose_bf16_k<<<DIM, DIM, 0, stream>>>(wq,  wqT);
  transpose_bf16_k<<<DIM, DIM, 0, stream>>>(wk,  wkT);
  transpose_bf16_k<<<DIM, DIM, 0, stream>>>(wv,  wvT);
  transpose_bf16_k<<<DIM, DIM, 0, stream>>>(pw2, pw2T);
  transpose_bf16_k<<<DIM, DIM, 0, stream>>>(aw1, aw1T);
  transpose_bf16_k<<<DIM, DIM, 0, stream>>>(aw2, aw2T);
  transpose_bf16_k<<<DIM, DIM, 0, stream>>>(fw,  fwT);

  // KNN
  knn_k<<<MTOT / 256, 256, 0, stream>>>(pos, knnw);

  // q/k/v projections: 8192 tiles of 16x16, 4 waves/block -> 2048 blocks
  const int gemm_blocks = (MTOT / 16) * (DIM / 16) / 4;
  gemm_wmma_bf16_k<<<gemm_blocks, 128, 0, stream>>>(xb, wqT, nullptr, nullptr, qb, nullptr);
  gemm_wmma_bf16_k<<<gemm_blocks, 128, 0, stream>>>(xb, wkT, nullptr, nullptr, kb, nullptr);
  gemm_wmma_bf16_k<<<gemm_blocks, 128, 0, stream>>>(xb, wvT, nullptr, nullptr, vb, nullptr);

  // Fused attention: 4 points per wave, one wave per block,
  // LDS = 3 x 32KB activations + 128KB TDM-staged weight + gather scratch
  const size_t smem_bytes =
      (size_t)(3 * PTS_PER_WAVE * KNN * DIM + DIM * DIM) * sizeof(bf16) +
      (size_t)PTS_PER_WAVE * KNN * 4 * sizeof(float) +
      (size_t)PTS_PER_WAVE * KNN * sizeof(int);
  attn_k<<<MTOT / PTS_PER_WAVE, 32, smem_bytes, stream>>>(
      pos, qb, kb, vb, knnw, pw1, pb1, pw2T, pb2, aw1T, ab1, aw2T, ab2, aggb);

  // Final linear + bias + residual -> f32 output
  gemm_wmma_bf16_k<<<gemm_blocks, 128, 0, stream>>>(aggb, fwT, fb, x, nullptr,
                                                    (float*)d_out);
}
